// AtomTransformer_85899345920321
// MI455X (gfx1250) — matrix operand order repaired and checked by reading, not compiled, not run on hardware
//
#include <hip/hip_runtime.h>
#include <cstdint>
#include <cstddef>

// ---------------- problem constants (match reference) ----------------
#define NA   2048   // atoms
#define CC   128    // channels
#define NQ   32     // queries per block
#define NK   128    // key window
#define NH   4      // heads
#define DH   32     // head dim
#define NJB  (NA / NQ)   // 64 query blocks

typedef __attribute__((ext_vector_type(16))) __bf16 v16bf;
typedef __attribute__((ext_vector_type(8)))  __bf16 v8bf;
typedef __attribute__((ext_vector_type(8)))  float  v8f;
typedef __attribute__((ext_vector_type(4)))  int    v4i;

__device__ __forceinline__ float sigf(float x) { return 1.f / (1.f + __expf(-x)); }

// gfx1250 async global->LDS staging (probe via __has_builtin; safe fallback)
#if defined(__HIP_DEVICE_COMPILE__)
#  if __has_builtin(__builtin_amdgcn_global_load_async_to_lds_b128)
#    define HAVE_ASYNC_LDS 1
#    define ASYNC_B128(gsrc, ldst)                                             \
        __builtin_amdgcn_global_load_async_to_lds_b128(                        \
            (v4i*)(gsrc), (v4i*)(ldst), 0, 0)
#    if __has_builtin(__builtin_amdgcn_s_wait_asynccnt)
#      define WAIT_ASYNC() __builtin_amdgcn_s_wait_asynccnt(0)
#    else
#      define WAIT_ASYNC() asm volatile("s_wait_asynccnt 0" ::: "memory")
#    endif
#  endif
#endif

// ---------------------------------------------------------------------
// Row LayerNorm over 128 channels: one wave32 per row, float4 per lane.
// ---------------------------------------------------------------------
__global__ void ln_rows128(const float* __restrict__ X, float* __restrict__ Y, int nrows)
{
    int row  = blockIdx.x * 8 + (threadIdx.x >> 5);   // 8 waves / block
    int lane = threadIdx.x & 31;
    if (row >= nrows) return;
    const float4 v = ((const float4*)(X + (size_t)row * CC))[lane];
    float s = v.x + v.y + v.z + v.w;
#pragma unroll
    for (int m = 16; m; m >>= 1) s += __shfl_xor(s, m, 32);
    float mu = s * (1.f / CC);
    float d0 = v.x - mu, d1 = v.y - mu, d2 = v.z - mu, d3 = v.w - mu;
    float q  = d0*d0 + d1*d1 + d2*d2 + d3*d3;
#pragma unroll
    for (int m = 16; m; m >>= 1) q += __shfl_xor(q, m, 32);
    float rstd = rsqrtf(q * (1.f / CC) + 1e-5f);
    float4 o; o.x = d0*rstd; o.y = d1*rstd; o.z = d2*rstd; o.w = d3*rstd;
    ((float4*)(Y + (size_t)row * CC))[lane] = o;
}

// ---------------- converters ----------------
__global__ void cvt_f32_bf16(__bf16* __restrict__ Y, const float* __restrict__ X, int n)
{ int i = blockIdx.x * 256 + threadIdx.x; if (i < n) Y[i] = (__bf16)X[i]; }

// W[K][N] f32 -> Wt[N][K] bf16 (transposed so B-fragments are contiguous)
__global__ void wt_cvt_t(__bf16* __restrict__ Wt, const float* __restrict__ W, int K, int N)
{
    int i = blockIdx.x * 256 + threadIdx.x;
    if (i < K * N) { int k = i / N, n = i - k * N; Wt[(size_t)n * K + k] = (__bf16)W[i]; }
}

// ---------------------------------------------------------------------
// GEMM: C[M,N] = A[M,K] @ Bt[N,K]^T (+ bias[N]); bf16 operands,
// v_wmma_f32_16x16x32_bf16, one wave per 16x16 tile, all-vector b128 loads.
// Output either f32 (Cf) or scaled bf16 (Cbf).
// ---------------------------------------------------------------------
__global__ void gemm_bf16t(const __bf16* __restrict__ A, const __bf16* __restrict__ Bt,
                           const float* __restrict__ bias, float* __restrict__ Cf,
                           __bf16* __restrict__ Cbf, float outScale,
                           int M, int N, int K)
{
    const int wave = threadIdx.x >> 5;
    const int lane = threadIdx.x & 31;
    const int l16  = lane & 15;
    const int half = lane >> 4;
    const int tilesN = N >> 4;
    const int tile = blockIdx.x * 8 + wave;           // grid sized exactly
    const int tm = (tile / tilesN) << 4;
    const int tn = (tile % tilesN) << 4;

    v8f acc = {0.f,0.f,0.f,0.f,0.f,0.f,0.f,0.f};
    const __bf16* Arow = A  + (size_t)(tm + l16) * K;
    const __bf16* Brow = Bt + (size_t)(tn + l16) * K;
    for (int k0 = 0; k0 < K; k0 += 32) {
        const int ka = k0 + (half << 3);              // A: lanes0-15 K=0..7,16..23 ; +8 for upper
        const v8bf a0 = *(const v8bf*)(Arow + ka);
        const v8bf a1 = *(const v8bf*)(Arow + ka + 16);
        const int kb = k0 + (half << 3);              // B: lanes0-15 K=0..7,16..23 ; +8 for upper
        const v8bf b0 = *(const v8bf*)(Brow + kb);
        const v8bf b1 = *(const v8bf*)(Brow + kb + 16);
        const v16bf af  = __builtin_shufflevector(a0, a1, 0,1,2,3,4,5,6,7,8,9,10,11,12,13,14,15);
        const v16bf bfv = __builtin_shufflevector(b0, b1, 0,1,2,3,4,5,6,7,8,9,10,11,12,13,14,15);
        acc = __builtin_amdgcn_wmma_f32_16x16x32_bf16(false, af, false, bfv,
                                                      (short)0, acc, false, false);
    }
    const float bv = bias ? bias[tn + l16] : 0.f;
    if (Cf) {
        float* Cp = Cf + (size_t)(tm + (half << 3)) * N + (tn + l16);
#pragma unroll
        for (int g = 0; g < 8; ++g) Cp[(size_t)g * N] = acc[g] + bv;
    } else {
        __bf16* Cp = Cbf + (size_t)(tm + (half << 3)) * N + (tn + l16);
#pragma unroll
        for (int g = 0; g < 8; ++g) Cp[(size_t)g * N] = (__bf16)((acc[g] + bv) * outScale);
    }
}

// ---------------------------------------------------------------------
// Pair bias over the sliding window only (~6% of z is ever touched):
//   bias[jb][h][q][k] = (LN(z[q,k,:16])*ln_z_w + ln_z_b) @ W_b[:,h]
// Out-of-range keys -> -1e30.
// ---------------------------------------------------------------------
__global__ void pair_bias_kernel(const float* __restrict__ z,
                                 const float* __restrict__ lnw,
                                 const float* __restrict__ lnb,
                                 const float* __restrict__ Wb,
                                 float* __restrict__ biasb)
{
    const int jb = blockIdx.x;
    const int kstart = 32 * jb - 48;
    __shared__ float sWb[16 * NH], sLw[16], sLb[16];
    if (threadIdx.x < 64) sWb[threadIdx.x] = Wb[threadIdx.x];
    if (threadIdx.x >= 64 && threadIdx.x < 80) sLw[threadIdx.x - 64] = lnw[threadIdx.x - 64];
    if (threadIdx.x >= 80 && threadIdx.x < 96) sLb[threadIdx.x - 80] = lnb[threadIdx.x - 80];
    __syncthreads();

    for (int p = threadIdx.x; p < NQ * NK; p += 256) {
        const int qi = p >> 7, kk = p & 127;
        const int k = kstart + kk;
        float o[NH];
        if (k >= 0 && k < NA) {
            const float* zp = z + ((size_t)(jb * NQ + qi) * NA + k) * 16;
            if (p + 256 < NQ * NK)                        // gfx1250 global_prefetch_b8
                __builtin_prefetch(zp + (size_t)2 * NA * 16, 0, 0);
            float v[16], mu = 0.f;
#pragma unroll
            for (int i = 0; i < 16; ++i) { v[i] = zp[i]; mu += v[i]; }
            mu *= (1.f / 16.f);
            float var = 0.f;
#pragma unroll
            for (int i = 0; i < 16; ++i) { float d = v[i] - mu; var += d * d; }
            const float rstd = rsqrtf(var * (1.f / 16.f) + 1e-5f);
#pragma unroll
            for (int hh = 0; hh < NH; ++hh) o[hh] = 0.f;
#pragma unroll
            for (int i = 0; i < 16; ++i) {
                const float y = (v[i] - mu) * rstd * sLw[i] + sLb[i];
#pragma unroll
                for (int hh = 0; hh < NH; ++hh) o[hh] += y * sWb[i * NH + hh];
            }
        } else {
#pragma unroll
            for (int hh = 0; hh < NH; ++hh) o[hh] = -1e30f;
        }
        const size_t base = ((size_t)jb * NH) * NQ * NK + (size_t)qi * NK + kk;
#pragma unroll
        for (int hh = 0; hh < NH; ++hh) biasb[base + (size_t)hh * NQ * NK] = o[hh];
    }
}

// ---------------------------------------------------------------------
// Block-local attention: one workgroup (4 waves) per (query block, head).
// Q already scaled by 1/sqrt(d) (folded into its GEMM epilogue).
// K/V/Q staged in LDS via async b128 loads where available.
// ---------------------------------------------------------------------
__global__ void attn_kernel(const __bf16* __restrict__ Qm, const __bf16* __restrict__ Km,
                            const __bf16* __restrict__ Vm, const float* __restrict__ biasb,
                            float* __restrict__ Om)
{
    const int jb = blockIdx.x >> 2;
    const int h  = blockIdx.x & 3;
    __shared__ alignas(16) __bf16 Qs[NQ * DH];
    __shared__ alignas(16) __bf16 Ks[NK * DH];
    __shared__ alignas(16) __bf16 Vs[NK * DH];
    __shared__ alignas(16) float  Ls[NQ * NK];
    __shared__ alignas(16) __bf16 Ps[NQ * NK];

    const int tid = threadIdx.x;
    const int kstart = 32 * jb - 48;
    const bool edge = (kstart < 0) || (kstart + NK > NA);   // jb 0,1,62,63

#ifdef HAVE_ASYNC_LDS
    {   // Q: 32 rows x 64B = 128 chunks of 16B, one per thread
        const int r = tid >> 2, sub = tid & 3;
        ASYNC_B128(Qm + (size_t)(jb * NQ + r) * CC + h * DH + sub * 8,
                   Qs + r * DH + sub * 8);
        if (!edge) {                                   // K/V: 8KB each, 4 rounds
#pragma unroll
            for (int it = 0; it < 4; ++it) {
                const int ci = it * 128 + tid;
                const int rr = ci >> 2, ss = ci & 3;
                const size_t g = (size_t)(kstart + rr) * CC + h * DH + ss * 8;
                ASYNC_B128(Km + g, Ks + rr * DH + ss * 8);
                ASYNC_B128(Vm + g, Vs + rr * DH + ss * 8);
            }
        }
        WAIT_ASYNC();
    }
    if (edge)
#else
    for (int i = tid; i < NQ * DH; i += 128) {
        const int r = i >> 5, c = i & 31;
        Qs[i] = Qm[(size_t)(jb * NQ + r) * CC + h * DH + c];
    }
#endif
    {   // range-checked fallback staging of K/V
        for (int i = tid; i < NK * DH; i += 128) {
            const int r = i >> 5, c = i & 31;
            const int krow = kstart + r;
            __bf16 kv = (__bf16)0.f, vv = (__bf16)0.f;
            if (krow >= 0 && krow < NA) {
                kv = Km[(size_t)krow * CC + h * DH + c];
                vv = Vm[(size_t)krow * CC + h * DH + c];
            }
            Ks[i] = kv; Vs[i] = vv;
        }
    }
    __syncthreads();

    const int wave = tid >> 5, lane = tid & 31;
    const int l16 = lane & 15, half = lane >> 4;
    const float* bp = biasb + (size_t)(jb * NH + h) * NQ * NK;

    // ---- logits = (Q*scale) @ K^T + bias : 2x8 tiles over 4 waves ----
    for (int t = wave; t < 16; t += 4) {
        const int mt = t >> 3, nt = t & 7;
        v16bf af, bfv;
        const int m  = mt * 16 + l16;
        const int ka = half * 16;
#pragma unroll
        for (int i = 0; i < 8; ++i) af[i] = Qs[m * DH + ka + i];
#pragma unroll
        for (int i = 0; i < 8; ++i) af[8 + i] = Qs[m * DH + ka + 8 + i];
        const int nn = nt * 16 + l16;                  // key index (B col)
        const int kb = half * 16;                      // head-dim  (B row)
#pragma unroll
        for (int i = 0; i < 16; ++i) bfv[i] = Ks[nn * DH + kb + i];   // B[kd][n]=Ks[n][kd]
        v8f acc = {0.f,0.f,0.f,0.f,0.f,0.f,0.f,0.f};
        acc = __builtin_amdgcn_wmma_f32_16x16x32_bf16(false, af, false, bfv,
                                                      (short)0, acc, false, false);
#pragma unroll
        for (int g = 0; g < 8; ++g) {
            const int row = mt * 16 + half * 8 + g;
            Ls[row * NK + nt * 16 + l16] = acc[g] + bp[row * NK + nt * 16 + l16];
        }
    }
    __syncthreads();

    // ---- row softmax (wave 0, one lane per query row) ----
    if (tid < NQ) {
        float mx = -3.0e38f;
        for (int kk = 0; kk < NK; ++kk) mx = fmaxf(mx, Ls[tid * NK + kk]);
        float sum = 0.f;
        for (int kk = 0; kk < NK; ++kk) {
            const float e = __expf(Ls[tid * NK + kk] - mx);
            Ls[tid * NK + kk] = e; sum += e;
        }
        const float inv = 1.f / sum;
        for (int kk = 0; kk < NK; ++kk) Ps[tid * NK + kk] = (__bf16)(Ls[tid * NK + kk] * inv);
    }
    __syncthreads();

    // ---- O = P @ V : 2x2 tiles, one per wave, K = 128 ----
    {
        const int mt = wave >> 1, nt = wave & 1;
        v8f acc = {0.f,0.f,0.f,0.f,0.f,0.f,0.f,0.f};
        for (int k0 = 0; k0 < NK; k0 += 32) {
            v16bf af, bfv;
            const int m  = mt * 16 + l16;
            const int ka = k0 + half * 16;
#pragma unroll
            for (int i = 0; i < 8; ++i) af[i] = Ps[m * NK + ka + i];
#pragma unroll
            for (int i = 0; i < 8; ++i) af[8 + i] = Ps[m * NK + ka + 8 + i];
            const int nn = nt * 16 + l16;
            const int kb = k0 + half * 16;
#pragma unroll
            for (int i = 0; i < 16; ++i) bfv[i] = Vs[(kb + i) * DH + nn];
            acc = __builtin_amdgcn_wmma_f32_16x16x32_bf16(false, af, false, bfv,
                                                          (short)0, acc, false, false);
        }
#pragma unroll
        for (int g = 0; g < 8; ++g) {
            const int row = jb * NQ + mt * 16 + half * 8 + g;
            Om[(size_t)row * CC + h * DH + nt * 16 + l16] = acc[g];
        }
    }
}

// ---------------- elementwise glue kernels ----------------
__global__ void ew_scale_cols_bf(__bf16* __restrict__ Y, const float* __restrict__ X,
                                 const float* __restrict__ w, int n, int C)
{ int i = blockIdx.x * 256 + threadIdx.x; if (i < n) Y[i] = (__bf16)(X[i] * w[i % C]); }

__global__ void ew_adaln_bf(__bf16* __restrict__ an, const float* __restrict__ G,
                            const float* __restrict__ bg, const float* __restrict__ aln,
                            const float* __restrict__ S, int n, int C)
{ int i = blockIdx.x * 256 + threadIdx.x;
  if (i < n) an[i] = (__bf16)(sigf(G[i] + bg[i % C]) * aln[i] + S[i]); }

__global__ void ew_sigmul_f32(float* __restrict__ out, const float* __restrict__ X,
                              const float* __restrict__ b, const float* __restrict__ Y,
                              int n, int C)
{ int i = blockIdx.x * 256 + threadIdx.x;
  if (i < n) { float bb = b ? b[i % C] : 0.f; out[i] = sigf(X[i] + bb) * Y[i]; } }

__global__ void ew_sigmul_bf(__bf16* __restrict__ out, const float* __restrict__ X,
                             const float* __restrict__ b, const float* __restrict__ Y,
                             int n, int C)
{ int i = blockIdx.x * 256 + threadIdx.x;
  if (i < n) { float bb = b ? b[i % C] : 0.f; out[i] = (__bf16)(sigf(X[i] + bb) * Y[i]); } }

__global__ void ew_sigmul_add(float* __restrict__ out, const float* __restrict__ X,
                              const float* __restrict__ b, const float* __restrict__ Y,
                              const float* __restrict__ Z, int n, int C)
{ int i = blockIdx.x * 256 + threadIdx.x;
  if (i < n) out[i] = sigf(X[i] + b[i % C]) * Y[i] + Z[i]; }

__global__ void ew_swiglu_bf(__bf16* __restrict__ out, const float* __restrict__ A,
                             const float* __restrict__ B, int n)
{ int i = blockIdx.x * 256 + threadIdx.x;
  if (i < n) { float x = A[i]; out[i] = (__bf16)(x * sigf(x) * B[i]); } }

// ---------------------------------------------------------------------
static inline void gemm(const __bf16* A, const __bf16* Bt, const float* bias,
                        float* Cf, __bf16* Cbf, float outScale,
                        int M, int N, int K, hipStream_t st)
{
    const int tiles = (M / 16) * (N / 16);             // always divisible by 8 here
    gemm_bf16t<<<tiles / 8, 256, 0, st>>>(A, Bt, bias, Cf, Cbf, outScale, M, N, K);
}

extern "C" void kernel_launch(void* const* d_in, const int* in_sizes, int n_in,
                              void* d_out, int out_size, void* d_ws, size_t ws_size,
                              hipStream_t stream)
{
    (void)in_sizes; (void)n_in; (void)out_size; (void)ws_size;

    const float* q_in = (const float*)d_in[0];
    const float* c_in = (const float*)d_in[1];
    const float* z_in = (const float*)d_in[2];

    char* wsb = (char*)d_ws;
    size_t off = 0;
    auto allocf = [&](size_t n) { float*  p = (float*)(wsb + off);  off += n * 4; return p; };
    auto allocb = [&](size_t n) { __bf16* p = (__bf16*)(wsb + off); off += n * 2; return p; };
    const size_t NC = (size_t)NA * CC;

    // f32 buffers
    float* a_cur = allocf(NC);
    float* a_ln  = allocf(NC);
    float* s_ln  = allocf(NC);
    float* Gbuf  = allocf(NC);
    float* Sbuf  = allocf(NC);
    float* Gt    = allocf(NC);
    float* Obuf  = allocf(NC);
    float* OUTb  = allocf(NC);
    float* SGr   = allocf(NC);
    float* b_out = allocf(NC);
    float* H1    = allocf(NC * 2);
    float* H2    = allocf(NC * 2);
    float* Tb    = allocf(NC);
    float* biasb = allocf((size_t)NJB * NH * NQ * NK);
    // bf16 buffers (GEMM operands / attention inputs)
    __bf16* s_nb = allocb(NC);
    __bf16* a_nb = allocb(NC);
    __bf16* cbf  = allocb(NC);
    __bf16* GOb  = allocb(NC);
    __bf16* Hsb  = allocb(NC * 2);
    __bf16* Qbf  = allocb(NC);
    __bf16* Kbf  = allocb(NC);
    __bf16* Vbf  = allocb(NC);
    // transposed bf16 weights (reused each block iteration)
    __bf16* Wt_g   = allocb(CC * CC);
    __bf16* Wt_s   = allocb(CC * CC);
    __bf16* Wt_q   = allocb(CC * CC);
    __bf16* Wt_k   = allocb(CC * CC);
    __bf16* Wt_v   = allocb(CC * CC);
    __bf16* Wt_gt  = allocb(CC * CC);
    __bf16* Wt_o   = allocb(CC * CC);
    __bf16* Wt_sg  = allocb(CC * CC);
    __bf16* Wt_tg  = allocb(CC * CC);
    __bf16* Wt_ts  = allocb(CC * CC);
    __bf16* Wt_tsg = allocb(CC * CC);
    __bf16* Wt_1   = allocb(CC * 2 * CC);
    __bf16* Wt_2   = allocb(CC * 2 * CC);
    __bf16* Wt_out = allocb(2 * CC * CC);

    (void)hipMemcpyAsync(a_cur, q_in, NC * sizeof(float), hipMemcpyDeviceToDevice, stream);
    ln_rows128<<<NA / 8, 256, 0, stream>>>(c_in, s_ln, NA);         // LN(c)
    cvt_f32_bf16<<<NC / 256, 256, 0, stream>>>(cbf, c_in, (int)NC); // raw c for W_sg gates

    const int nEW = (int)NC, gEW = nEW / 256;
    const int nEW2 = nEW * 2, gEW2 = gEW * 2;
    const float QSCALE = 0.17677669529663687f;         // 1/sqrt(32)
    auto cvtw = [&](const float* W, __bf16* Wt, int K, int N) {
        wt_cvt_t<<<(K * N) / 256, 256, 0, stream>>>(Wt, W, K, N);
    };

    for (int bk = 0; bk < 3; ++bk) {
        // JAX tree-flatten order (dict keys sorted): per block 15 apb + 9 tr leaves.
        const float* const* P = (const float* const*)(d_in + 3 + bk * 24);
        const float *W_b = P[0],  *W_g = P[1],  *W_gate = P[2], *W_k = P[3],  *W_o = P[4];
        const float *W_q = P[5],  *W_s = P[6],  *W_sg = P[7],   *W_v = P[8];
        const float *b_g = P[9],  *b_q = P[10], *b_sg = P[11];
        const float *ln_s_w = P[12], *ln_z_b = P[13], *ln_z_w = P[14];
        const float *tW_1 = P[15], *tW_2 = P[16], *tW_g = P[17], *tW_out = P[18];
        const float *tW_s = P[19], *tW_sg = P[20], *tb_g = P[21], *tb_sg = P[22];
        const float *tln_s_w = P[23];

        // weight prep (bf16 + transpose)
        cvtw(W_g, Wt_g, CC, CC);     cvtw(W_s, Wt_s, CC, CC);
        cvtw(W_q, Wt_q, CC, CC);     cvtw(W_k, Wt_k, CC, CC);
        cvtw(W_v, Wt_v, CC, CC);     cvtw(W_gate, Wt_gt, CC, CC);
        cvtw(W_o, Wt_o, CC, CC);     cvtw(W_sg, Wt_sg, CC, CC);
        cvtw(tW_g, Wt_tg, CC, CC);   cvtw(tW_s, Wt_ts, CC, CC);
        cvtw(tW_sg, Wt_tsg, CC, CC); cvtw(tW_1, Wt_1, CC, 2 * CC);
        cvtw(tW_2, Wt_2, CC, 2 * CC); cvtw(tW_out, Wt_out, 2 * CC, CC);

        // ---------------- attention with pair bias ----------------
        ln_rows128<<<NA / 8, 256, 0, stream>>>(a_cur, a_ln, NA);
        ew_scale_cols_bf<<<gEW, 256, 0, stream>>>(s_nb, s_ln, ln_s_w, nEW, CC);
        gemm(s_nb, Wt_g, nullptr, Gbuf, nullptr, 1.f, NA, CC, CC, stream);
        gemm(s_nb, Wt_s, nullptr, Sbuf, nullptr, 1.f, NA, CC, CC, stream);
        ew_adaln_bf<<<gEW, 256, 0, stream>>>(a_nb, Gbuf, b_g, a_ln, Sbuf, nEW, CC);
        gemm(a_nb, Wt_q, b_q,    nullptr, Qbf, QSCALE, NA, CC, CC, stream);
        gemm(a_nb, Wt_k, nullptr, nullptr, Kbf, 1.f,   NA, CC, CC, stream);
        gemm(a_nb, Wt_v, nullptr, nullptr, Vbf, 1.f,   NA, CC, CC, stream);
        gemm(a_nb, Wt_gt, nullptr, Gt, nullptr, 1.f,   NA, CC, CC, stream);
        pair_bias_kernel<<<NJB, 256, 0, stream>>>(z_in, ln_z_w, ln_z_b, W_b, biasb);
        attn_kernel<<<NJB * NH, 128, 0, stream>>>(Qbf, Kbf, Vbf, biasb, Obuf);
        ew_sigmul_bf<<<gEW, 256, 0, stream>>>(GOb, Gt, nullptr, Obuf, nEW, CC);
        gemm(GOb, Wt_o, nullptr, OUTb, nullptr, 1.f, NA, CC, CC, stream);
        gemm(cbf, Wt_sg, nullptr, SGr, nullptr, 1.f, NA, CC, CC, stream);
        ew_sigmul_f32<<<gEW, 256, 0, stream>>>(b_out, SGr, b_sg, OUTb, nEW, CC);

        // ---------------- transition (uses pre-block a) ----------------
        ew_scale_cols_bf<<<gEW, 256, 0, stream>>>(s_nb, s_ln, tln_s_w, nEW, CC);
        gemm(s_nb, Wt_tg, nullptr, Gbuf, nullptr, 1.f, NA, CC, CC, stream);
        gemm(s_nb, Wt_ts, nullptr, Sbuf, nullptr, 1.f, NA, CC, CC, stream);
        ew_adaln_bf<<<gEW, 256, 0, stream>>>(a_nb, Gbuf, tb_g, a_ln, Sbuf, nEW, CC);
        gemm(a_nb, Wt_1, nullptr, H1, nullptr, 1.f, NA, 2 * CC, CC, stream);
        gemm(a_nb, Wt_2, nullptr, H2, nullptr, 1.f, NA, 2 * CC, CC, stream);
        ew_swiglu_bf<<<gEW2, 256, 0, stream>>>(Hsb, H1, H2, nEW2);
        gemm(Hsb, Wt_out, nullptr, Tb, nullptr, 1.f, NA, CC, 2 * CC, stream);
        gemm(cbf, Wt_tsg, nullptr, SGr, nullptr, 1.f, NA, CC, CC, stream);

        float* dst = (bk == 2) ? (float*)d_out : a_cur;
        ew_sigmul_add<<<gEW, 256, 0, stream>>>(dst, SGr, tb_sg, Tb, b_out, nEW, CC);
    }
}
